// Qwen3_5FullAttention_10118942949427
// MI455X (gfx1250) — matrix operand order repaired and checked by reading, not compiled, not run on hardware
//
#include <hip/hip_runtime.h>
#include <hip/hip_bf16.h>
#include <stdint.h>

// ---------------------------------------------------------------------------
// Qwen3.5 full-attention block for MI455X (gfx1250), wave32 + WMMA bf16 + TDM.
// T=2048, HIDDEN=2048, HQ=16, HKV=4, D=128.
// ---------------------------------------------------------------------------

typedef __attribute__((ext_vector_type(16))) __bf16 bf16x16;
typedef __attribute__((ext_vector_type(8)))  float  f32x8;
typedef __attribute__((ext_vector_type(4)))  unsigned int u32x4;
typedef __attribute__((ext_vector_type(8)))  int i32x8;
typedef __attribute__((ext_vector_type(4)))  int i32x4;

union FragAB {
    bf16x16 v;
    uint4   q[2];
};

__device__ inline unsigned short f32_to_bf16_raw(float f) {
    union { float f; uint32_t u; } v; v.f = f;
    uint32_t u = v.u;
    uint32_t r = u + 0x7FFFu + ((u >> 16) & 1u);   // round-to-nearest-even
    return (unsigned short)(r >> 16);
}

// ---------------------------------------------------------------------------
// Tensor Data Mover (gfx1250): DMA a 2-D bf16 tile (tileRows x 32) from global
// into LDS with 16B padding after every 64B row -> 80B LDS row stride.
// Descriptor per CDNA5 ISA.
// This toolchain's builtin takes 6 args: (g0 u32x4, g1 i32x8, g2 i32x4,
// g3 i32x4, g4 i32x8, cpol).
// ---------------------------------------------------------------------------
#if defined(__gfx1250__) && __has_builtin(__builtin_amdgcn_tensor_load_to_lds) && \
    __has_builtin(__builtin_amdgcn_s_wait_tensorcnt)
#define USE_TDM 1
#else
#define USE_TDM 0
#endif

#if USE_TDM
__device__ inline void tdm_load_tile_2d(const unsigned short* gsrc, unsigned lds_off,
                                        unsigned rowLen /*elements*/, unsigned tileRows) {
    const unsigned long long ga = (unsigned long long)(uintptr_t)gsrc;
    u32x4 g0;
    g0[0] = 0x1u;                                    // count=1, user descriptor
    g0[1] = lds_off;                                 // lds_addr (bytes)
    g0[2] = (unsigned)ga;                            // global_addr[31:0]
    g0[3] = ((unsigned)(ga >> 32) & 0x01FFFFFFu)     // global_addr[56:32]
            | (2u << 30);                            // type=2 ("image")
    i32x8 g1;
    g1[0] = (int)((1u << 16)      // data_size = 1 -> 2-byte elements
                | (1u << 20)      // pad_enable
                | (3u << 22)      // pad_interval: pad after 16 DWORDs (64 B)
                | (3u << 25));    // pad_amount:   4 DWORDs (16 B)
    g1[1] = (int)((rowLen & 0xFFFFu) << 16);                          // tensor_dim0 lo16
    g1[2] = (int)(((rowLen >> 16) & 0xFFFFu)
                | ((tileRows & 0xFFFFu) << 16));                      // dim0 hi | tensor_dim1 lo
    g1[3] = (int)(((tileRows >> 16) & 0xFFFFu) | (32u << 16));        // dim1 hi | tile_dim0=32
    g1[4] = (int)(tileRows & 0xFFFFu);                                // tile_dim1 | tile_dim2=0
    g1[5] = (int)rowLen;                                              // tensor_dim0_stride lo32
    g1[6] = 0;                                                        // stride hi | dim1_stride lo
    g1[7] = 0;
    const i32x4 z4 = {0, 0, 0, 0};
    const i32x8 z8 = {0, 0, 0, 0, 0, 0, 0, 0};
    __builtin_amdgcn_tensor_load_to_lds(g0, g1, z4, z4, z8, 0);
}
#endif

// ---------------------------------------------------------------------------
// Elementwise f32 -> bf16 (row-major preserved)
// ---------------------------------------------------------------------------
__global__ __launch_bounds__(256) void conv_bf16_kernel(
    const float* __restrict__ in, unsigned short* __restrict__ out, size_t n) {
    size_t i = (size_t)blockIdx.x * 256 + threadIdx.x;
    if (i < n) out[i] = f32_to_bf16_raw(in[i]);
}

// ---------------------------------------------------------------------------
// f32 [K,N] row-major -> bf16 [N,K] row-major (weight transpose for WMMA B)
// ---------------------------------------------------------------------------
__global__ __launch_bounds__(256) void transpose_bf16_kernel(
    const float* __restrict__ in, unsigned short* __restrict__ out, int K, int N) {
    __shared__ float tile[32][33];
    const int k0 = blockIdx.y * 32, n0 = blockIdx.x * 32;
    const int c = threadIdx.x & 31, r0 = threadIdx.x >> 5;   // r0: 0..7
#pragma unroll
    for (int i = 0; i < 4; ++i)
        tile[r0 + 8 * i][c] = in[(size_t)(k0 + r0 + 8 * i) * N + n0 + c];
    __syncthreads();
#pragma unroll
    for (int i = 0; i < 4; ++i)
        out[(size_t)(n0 + r0 + 8 * i) * K + k0 + c] = f32_to_bf16_raw(tile[c][r0 + 8 * i]);
}

// ---------------------------------------------------------------------------
// C[M,N] = A[M,K] x B^T  (A bf16 row-major, B given as [N,K] bf16 row-major)
// Block: 256 thr = 8 waves; block tile 128x128; wave tile 64x32 (4x2 frags).
// LDS rows padded to 40 shorts (80 B) -> conflict-free ds_load_b128.
// TDM path: double-buffered tensor_load_to_lds (waves 0/1 drive the DMA),
// one barrier per K-step. Fallback: manual b128 staging, two barriers.
// ---------------------------------------------------------------------------
__global__ __launch_bounds__(256) void gemm_bf16_kernel(
    const unsigned short* __restrict__ A,
    const unsigned short* __restrict__ B,
    float* __restrict__ C,
    int M, int N, int K)
{
    __shared__ unsigned short As[2][128 * 40];
    __shared__ unsigned short Bs[2][128 * 40];
    const int tid  = threadIdx.x;
    const int m0   = blockIdx.y * 128;
    const int n0   = blockIdx.x * 128;
    const int w    = tid >> 5;
    const int lane = tid & 31;
    const int lh   = lane >> 4;        // K-half selector
    const int ln   = lane & 15;        // row/col within fragment
    const int wm   = (w >> 2) * 64;    // wave M offset in tile
    const int wn   = (w & 3) * 32;     // wave N offset in tile

    f32x8 acc[4][2];
#pragma unroll
    for (int i = 0; i < 4; ++i)
#pragma unroll
        for (int j = 0; j < 2; ++j)
#pragma unroll
            for (int e = 0; e < 8; ++e) acc[i][j][e] = 0.0f;

#if USE_TDM
    const unsigned ldsA[2] = { (unsigned)(uintptr_t)&As[0][0], (unsigned)(uintptr_t)&As[1][0] };
    const unsigned ldsB[2] = { (unsigned)(uintptr_t)&Bs[0][0], (unsigned)(uintptr_t)&Bs[1][0] };
    if (w == 0) tdm_load_tile_2d(A + (size_t)m0 * K, ldsA[0], (unsigned)K, 128u);
    if (w == 1) tdm_load_tile_2d(B + (size_t)n0 * K, ldsB[0], (unsigned)K, 128u);
    if (w < 2) __builtin_amdgcn_s_wait_tensorcnt(0);
    __syncthreads();
#endif

    for (int kk = 0; kk < K; kk += 32) {
        const int cur = (kk >> 5) & 1;
#if USE_TDM
        if (kk + 32 < K) {   // prefetch next tiles into the other buffer via TDM
            const int nxt = cur ^ 1;
            if (w == 0) tdm_load_tile_2d(A + (size_t)m0 * K + kk + 32, ldsA[nxt], (unsigned)K, 128u);
            if (w == 1) tdm_load_tile_2d(B + (size_t)n0 * K + kk + 32, ldsB[nxt], (unsigned)K, 128u);
        }
#else
        // cooperative stage: 128 rows x 32 bf16 for both A and B tiles
#pragma unroll
        for (int it = 0; it < 2; ++it) {
            int cid  = tid + it * 256;        // 0..511
            int row  = cid >> 2;
            int part = (cid & 3) * 8;         // short offset 0/8/16/24
            *reinterpret_cast<uint4*>(&As[cur][row * 40 + part]) =
                *reinterpret_cast<const uint4*>(A + (size_t)(m0 + row) * K + kk + part);
            *reinterpret_cast<uint4*>(&Bs[cur][row * 40 + part]) =
                *reinterpret_cast<const uint4*>(B + (size_t)(n0 + row) * K + kk + part);
        }
        __syncthreads();
#endif

        FragAB af[4], bf[2];
#pragma unroll
        for (int i = 0; i < 4; ++i) {
            const unsigned short* p = &As[cur][(wm + i * 16 + ln) * 40 + lh * 8];
            af[i].q[0] = *reinterpret_cast<const uint4*>(p);
            af[i].q[1] = *reinterpret_cast<const uint4*>(p + 16);
        }
#pragma unroll
        for (int j = 0; j < 2; ++j) {
            const unsigned short* p = &Bs[cur][(wn + j * 16 + ln) * 40 + lh * 8];
            bf[j].q[0] = *reinterpret_cast<const uint4*>(p);
            bf[j].q[1] = *reinterpret_cast<const uint4*>(p + 16);
        }
#pragma unroll
        for (int i = 0; i < 4; ++i)
#pragma unroll
            for (int j = 0; j < 2; ++j)
                acc[i][j] = __builtin_amdgcn_wmma_f32_16x16x32_bf16(
                    false, af[i].v, false, bf[j].v, (short)0, acc[i][j], false, false);

#if USE_TDM
        if (w < 2) __builtin_amdgcn_s_wait_tensorcnt(0);
        __syncthreads();   // also emits s_wait_dscnt 0: frag reads done before refill
#else
        __syncthreads();
#endif
    }

#pragma unroll
    for (int i = 0; i < 4; ++i)
#pragma unroll
        for (int j = 0; j < 2; ++j) {
            int col = n0 + wn + j * 16 + ln;
#pragma unroll
            for (int r = 0; r < 8; ++r) {
                int row = m0 + wm + i * 16 + r + 8 * lh;
                C[(size_t)row * N + col] = acc[i][j][r];
            }
        }
}

// ---------------------------------------------------------------------------
// Gemma RMSNorm + neox RoPE, one wave per (token, head).
// ---------------------------------------------------------------------------
__global__ __launch_bounds__(256) void qk_norm_rope_kernel(
    const int* __restrict__ positions,
    const float* __restrict__ proj,
    const float* __restrict__ normw,
    unsigned short* __restrict__ outb,
    float* __restrict__ outf,
    int nh, int rowStride, int headStride)
{
    const int gw   = blockIdx.x * 8 + (threadIdx.x >> 5);
    const int t    = gw / nh;
    const int h    = gw - t * nh;
    const int lane = threadIdx.x & 31;

    const float* src = proj + (size_t)t * rowStride + h * headStride;
    float x[4];
    float ss = 0.f;
#pragma unroll
    for (int i = 0; i < 4; ++i) { x[i] = src[lane + 32 * i]; ss += x[i] * x[i]; }
#pragma unroll
    for (int m = 16; m >= 1; m >>= 1) ss += __shfl_xor(ss, m, 32);
    const float inv = rsqrtf(ss * (1.0f / 128.0f) + 1e-6f);
#pragma unroll
    for (int i = 0; i < 4; ++i) x[i] *= inv * (1.0f + normw[lane + 32 * i]);

    const float pos = (float)positions[t];
    const float kf = -19.931568569324174f / 64.0f;   // -log2(1e6)/64
    float o[4];
    {
        float a0 = pos * exp2f((float)lane * kf);
        float a1 = pos * exp2f((float)(lane + 32) * kf);
        float c0, s0, c1, s1;
        __sincosf(a0, &s0, &c0);
        __sincosf(a1, &s1, &c1);
        o[0] = x[0] * c0 - x[2] * s0;
        o[2] = x[2] * c0 + x[0] * s0;
        o[1] = x[1] * c1 - x[3] * s1;
        o[3] = x[3] * c1 + x[1] * s1;
    }
    unsigned short* dst = outb + (size_t)t * (nh * 128) + h * 128;
#pragma unroll
    for (int i = 0; i < 4; ++i) dst[lane + 32 * i] = f32_to_bf16_raw(o[i]);
    if (outf) {
        float* df = outf + (size_t)t * (nh * 256) + h * 256;
#pragma unroll
        for (int i = 0; i < 4; ++i) df[lane + 32 * i] = o[i];
    }
}

// ---------------------------------------------------------------------------
// V: write transposed bf16 [kv*128+d][T] (WMMA-B friendly) + kv_fused f32 slot
// ---------------------------------------------------------------------------
__global__ __launch_bounds__(256) void v_prep_kernel(
    const float* __restrict__ vproj, unsigned short* __restrict__ vt,
    float* __restrict__ outf, int T)
{
    int idx = blockIdx.x * 256 + threadIdx.x;   // T*512
    int t = idx >> 9, c = idx & 511;
    int kv = c >> 7, d = c & 127;
    float v = vproj[idx];
    vt[(size_t)(kv * 128 + d) * T + t] = f32_to_bf16_raw(v);
    outf[(size_t)t * 1024 + kv * 256 + 128 + d] = v;
}

// ---------------------------------------------------------------------------
// Flash attention, WMMA bf16. One wave = 16 queries x D=128 for one head.
// ---------------------------------------------------------------------------
__global__ __launch_bounds__(256) void attn_kernel(
    const unsigned short* __restrict__ qr,   // bf16 [T, 2048]
    const unsigned short* __restrict__ krb,  // bf16 [T, 512]
    const unsigned short* __restrict__ vt,   // bf16 [512, T] (V transposed)
    const float* __restrict__ qg,            // f32 [T, 4096], gates at h*256+128
    unsigned short* __restrict__ att,        // bf16 [T, 2048]
    int T)
{
    __shared__ unsigned short Plds[8 * 16 * 40];
    const int tid  = threadIdx.x;
    const int w    = tid >> 5, lane = tid & 31;
    const int lh   = lane >> 4, ln = lane & 15;
    const int h    = blockIdx.y;
    const int kv   = h >> 2;
    const int qt0  = blockIdx.x * 128 + w * 16;
    unsigned short* P = &Plds[w * 16 * 40];

    FragAB qf[4];
    {
        const unsigned short* qrow = qr + (size_t)(qt0 + ln) * 2048 + h * 128;
#pragma unroll
        for (int c = 0; c < 4; ++c) {
            const unsigned short* p = qrow + c * 32 + lh * 8;
            qf[c].q[0] = *reinterpret_cast<const uint4*>(p);
            qf[c].q[1] = *reinterpret_cast<const uint4*>(p + 16);
        }
    }

    f32x8 acc[8];
    float m[8], l[8];
#pragma unroll
    for (int n = 0; n < 8; ++n)
#pragma unroll
        for (int r = 0; r < 8; ++r) acc[n][r] = 0.0f;
#pragma unroll
    for (int r = 0; r < 8; ++r) { m[r] = -1e30f; l[r] = 0.0f; }

    const float scale = 0.08838834764831845f;  // 1/sqrt(128)
    const float L2E   = 1.4426950408889634f;
    const int ntiles  = (qt0 + 47) >> 5;

    for (int kt = 0; kt < ntiles; ++kt) {
        const int kb = kt * 32;
        if (kt + 1 < ntiles) {  // global_prefetch next K/V tiles (L2-resident)
            __builtin_prefetch(krb + (size_t)(kb + 32 + ln) * 512 + kv * 128, 0, 1);
            __builtin_prefetch(vt + (size_t)(kv * 128 + ln * 8 + lh * 4) * T + kb + 32, 0, 1);
        }
        f32x8 s0, s1;
#pragma unroll
        for (int r = 0; r < 8; ++r) { s0[r] = 0.0f; s1[r] = 0.0f; }

#pragma unroll
        for (int c = 0; c < 4; ++c) {
            FragAB b0, b1;
            const unsigned short* k0p =
                krb + (size_t)(kb + ln) * 512 + kv * 128 + c * 32 + lh * 8;
            b0.q[0] = *reinterpret_cast<const uint4*>(k0p);
            b0.q[1] = *reinterpret_cast<const uint4*>(k0p + 16);
            const unsigned short* k1p =
                krb + (size_t)(kb + 16 + ln) * 512 + kv * 128 + c * 32 + lh * 8;
            b1.q[0] = *reinterpret_cast<const uint4*>(k1p);
            b1.q[1] = *reinterpret_cast<const uint4*>(k1p + 16);
            s0 = __builtin_amdgcn_wmma_f32_16x16x32_bf16(
                false, qf[c].v, false, b0.v, (short)0, s0, false, false);
            s1 = __builtin_amdgcn_wmma_f32_16x16x32_bf16(
                false, qf[c].v, false, b1.v, (short)0, s1, false, false);
        }

#pragma unroll
        for (int r = 0; r < 8; ++r) {
            const int qrow = qt0 + r + 8 * lh;
            float v0 = s0[r] * scale, v1 = s1[r] * scale;
            if (kb + ln > qrow)       v0 = -1e30f;
            if (kb + 16 + ln > qrow)  v1 = -1e30f;
            float mx = fmaxf(v0, v1);
#pragma unroll
            for (int msk = 8; msk >= 1; msk >>= 1) mx = fmaxf(mx, __shfl_xor(mx, msk, 32));
            const float mnew = fmaxf(m[r], mx);
            const float corr = exp2f((m[r] - mnew) * L2E);
            const float p0 = exp2f((v0 - mnew) * L2E);
            const float p1 = exp2f((v1 - mnew) * L2E);
            float ps = p0 + p1;
#pragma unroll
            for (int msk = 8; msk >= 1; msk >>= 1) ps += __shfl_xor(ps, msk, 32);
            l[r] = l[r] * corr + ps;
            m[r] = mnew;
#pragma unroll
            for (int n = 0; n < 8; ++n) acc[n][r] *= corr;
            P[(r + 8 * lh) * 40 + ln]      = f32_to_bf16_raw(p0);
            P[(r + 8 * lh) * 40 + 16 + ln] = f32_to_bf16_raw(p1);
        }

        FragAB pf;
        {
            const unsigned short* pp = &P[ln * 40 + lh * 8];
            pf.q[0] = *reinterpret_cast<const uint4*>(pp);
            pf.q[1] = *reinterpret_cast<const uint4*>(pp + 16);
        }

#pragma unroll
        for (int n = 0; n < 8; ++n) {
            FragAB vf;
            const unsigned short* vp =
                vt + (size_t)(kv * 128 + n * 16 + ln) * T + kb + lh * 8;
            vf.q[0] = *reinterpret_cast<const uint4*>(vp);
            vf.q[1] = *reinterpret_cast<const uint4*>(vp + 16);
            acc[n] = __builtin_amdgcn_wmma_f32_16x16x32_bf16(
                false, pf.v, false, vf.v, (short)0, acc[n], false, false);
        }
    }

#pragma unroll
    for (int n = 0; n < 8; ++n) {
#pragma unroll
        for (int r = 0; r < 8; ++r) {
            const int t = qt0 + r + 8 * lh;
            const int d = n * 16 + ln;
            const float g  = qg[(size_t)t * 4096 + h * 256 + 128 + d];
            const float ov = (acc[n][r] / l[r]) * (1.0f / (1.0f + expf(-g)));
            att[(size_t)t * 2048 + h * 128 + d] = f32_to_bf16_raw(ov);
        }
    }
}

// ---------------------------------------------------------------------------
// Host-side launch sequencing
// ---------------------------------------------------------------------------
extern "C" void kernel_launch(void* const* d_in, const int* in_sizes, int n_in,
                              void* d_out, int out_size, void* d_ws, size_t ws_size,
                              hipStream_t stream) {
    (void)in_sizes; (void)n_in; (void)out_size; (void)ws_size;

    const int*   positions = (const int*)d_in[0];
    const float* hs        = (const float*)d_in[1];
    const float* Wq        = (const float*)d_in[2];
    const float* Wk        = (const float*)d_in[3];
    const float* Wv        = (const float*)d_in[4];
    const float* Wo        = (const float*)d_in[5];
    const float* qnw       = (const float*)d_in[6];
    const float* knw       = (const float*)d_in[7];

    float* out  = (float*)d_out;               // [2048, 2048]
    float* out2 = out + (size_t)2048 * 2048;   // kv_fused [2048, 4, 256]

    char* ws = (char*)d_ws;
    size_t off = 0;
    unsigned short* hsb   = (unsigned short*)(ws + off); off += (size_t)2048 * 2048 * 2;
    unsigned short* WqT   = (unsigned short*)(ws + off); off += (size_t)4096 * 2048 * 2;
    unsigned short* WkT   = (unsigned short*)(ws + off); off += (size_t)512  * 2048 * 2;
    unsigned short* WvT   = (unsigned short*)(ws + off); off += (size_t)512  * 2048 * 2;
    unsigned short* WoT   = (unsigned short*)(ws + off); off += (size_t)2048 * 2048 * 2;
    float*          qgf   = (float*)(ws + off);          off += (size_t)2048 * 4096 * 4;
    float*          kproj = (float*)(ws + off);          off += (size_t)2048 * 512  * 4;
    float*          vproj = (float*)(ws + off);          off += (size_t)2048 * 512  * 4;
    unsigned short* qrb   = (unsigned short*)(ws + off); off += (size_t)2048 * 2048 * 2;
    unsigned short* krb   = (unsigned short*)(ws + off); off += (size_t)2048 * 512  * 2;
    unsigned short* vtb   = (unsigned short*)(ws + off); off += (size_t)512  * 2048 * 2;
    unsigned short* attb  = (unsigned short*)(ws + off); off += (size_t)2048 * 2048 * 2;

    conv_bf16_kernel<<<(2048 * 2048) / 256, 256, 0, stream>>>(hs, hsb, (size_t)2048 * 2048);
    transpose_bf16_kernel<<<dim3(4096 / 32, 2048 / 32), 256, 0, stream>>>(Wq, WqT, 2048, 4096);
    transpose_bf16_kernel<<<dim3(512  / 32, 2048 / 32), 256, 0, stream>>>(Wk, WkT, 2048, 512);
    transpose_bf16_kernel<<<dim3(512  / 32, 2048 / 32), 256, 0, stream>>>(Wv, WvT, 2048, 512);
    transpose_bf16_kernel<<<dim3(2048 / 32, 2048 / 32), 256, 0, stream>>>(Wo, WoT, 2048, 2048);
    gemm_bf16_kernel<<<dim3(4096 / 128, 2048 / 128), 256, 0, stream>>>(hsb, WqT, qgf,   2048, 4096, 2048);
    gemm_bf16_kernel<<<dim3(512  / 128, 2048 / 128), 256, 0, stream>>>(hsb, WkT, kproj, 2048, 512,  2048);
    gemm_bf16_kernel<<<dim3(512  / 128, 2048 / 128), 256, 0, stream>>>(hsb, WvT, vproj, 2048, 512,  2048);
    qk_norm_rope_kernel<<<(2048 * 16) / 8, 256, 0, stream>>>(positions, qgf,   qnw, qrb, nullptr, 16, 4096, 256);
    qk_norm_rope_kernel<<<(2048 * 4)  / 8, 256, 0, stream>>>(positions, kproj, knw, krb, out2,    4,  512,  128);
    v_prep_kernel<<<(2048 * 512) / 256, 256, 0, stream>>>(vproj, vtb, out2, 2048);
    attn_kernel<<<dim3(2048 / 128, 16), 256, 0, stream>>>(qrb, krb, vtb, qgf, attb, 2048);
    gemm_bf16_kernel<<<dim3(2048 / 128, 2048 / 128), 256, 0, stream>>>(attb, WoT, out, 2048, 2048, 2048);
}
